// GRUModel_19902878449936
// MI455X (gfx1250) — matrix operand order repaired and checked by reading, not compiled, not run on hardware
//
#include <hip/hip_runtime.h>

// GRU (I=2, H=32), B=8192, T=1024 — transposed WMMA formulation.
// One wave32 per 16-row batch tile. State h kept as h^T in registers
// ([hidden][batch] D-layout). Per step:
//   gx^T  = w_ih x^T + bias      -> 6x v_wmma_f32_16x16x4_f32 (A=w_ih const)
//   pre^T = w_hh h^T + gx^T      -> 6x v_wmma_f32_16x16x32_f16 (A=w_hh const)
// No LDS; the only cross-lane op is an 8-value half-wave exchange to rebuild
// the f16 B operand (h^T) each step.

typedef __attribute__((ext_vector_type(16))) _Float16 v16h;
typedef __attribute__((ext_vector_type(8)))  float    v8f;
typedef __attribute__((ext_vector_type(2)))  float    v2f;

#define GRU_H 32
#define GRU_I 2

__device__ __forceinline__ float fast_sigmoid(float v) {
    float e = __builtin_amdgcn_exp2f(v * -1.44269504088896340736f);
    return __builtin_amdgcn_rcpf(1.0f + e);
}

__device__ __forceinline__ float fast_tanh(float v) {
    // tanh(v) = 2*sigmoid(2v) - 1
    float e = __builtin_amdgcn_exp2f(v * -2.88539008177792681472f);
    return __builtin_amdgcn_rcpf(1.0f + e) * 2.0f - 1.0f;
}

__global__ __launch_bounds__(32)
void gru_wmma_kernel(const float* __restrict__ x,     // [B, T, 2]
                     const float* __restrict__ w_ih,  // [96, 2]
                     const float* __restrict__ w_hh,  // [96, 32]
                     const float* __restrict__ b_ih,  // [96]
                     const float* __restrict__ b_hh,  // [96]
                     const float* __restrict__ fc_w,  // [1, 32]
                     const float* __restrict__ fc_b,  // [1]
                     float* __restrict__ out,         // [B, 1]
                     int T)
{
    const int lane = threadIdx.x;   // 0..31
    const int n16  = lane & 15;     // batch column within tile / gate row
    const int hi   = lane >> 4;     // half select
    const int b0   = blockIdx.x * 16;

    // ---- Constant A operands: w_ih tiles, f32 16x4 A-layout (K=2,3 zero) ----
    // lanes 0-15: M = n16, VGPR0 = K0, VGPR1 = K1; lanes 16-31: K2,K3 = 0
    v2f Aih[6];
    #pragma unroll
    for (int nt = 0; nt < 6; ++nt) {
        int g = nt * 16 + n16;
        v2f a;
        a[0] = hi ? 0.0f : w_ih[g * GRU_I + 0];
        a[1] = hi ? 0.0f : w_ih[g * GRU_I + 1];
        Aih[nt] = a;
    }

    // ---- Constant A operands: w_hh tiles, f16 16x32 A-layout ----
    // lane: M = n16; elems 0..7 -> K = 16*hi + i; elems 8..15 -> K = 16*hi + 8 + i
    v16h Ahh[6];
    #pragma unroll
    for (int nt = 0; nt < 6; ++nt) {
        const float* src = w_hh + (nt * 16 + n16) * GRU_H;
        v16h tmp;
        #pragma unroll
        for (int i = 0; i < 8; ++i) {
            tmp[i]     = (_Float16)src[16 * hi + i];
            tmp[8 + i] = (_Float16)src[16 * hi + 8 + i];
        }
        Ahh[nt] = tmp;
    }

    // ---- Bias tiles in C/D layout: elem j -> gate g = nt*16 + j + 8*hi ----
    v8f biasRZ[4], biasXn[2], biasHn[2];
    #pragma unroll
    for (int nt = 0; nt < 4; ++nt) {
        #pragma unroll
        for (int j = 0; j < 8; ++j) {
            int g = nt * 16 + j + 8 * hi;
            biasRZ[nt][j] = b_ih[g] + b_hh[g];
        }
    }
    #pragma unroll
    for (int nt = 0; nt < 2; ++nt) {
        #pragma unroll
        for (int j = 0; j < 8; ++j) {
            int g = 64 + nt * 16 + j + 8 * hi;
            biasXn[nt][j] = b_ih[g];
            biasHn[nt][j] = b_hh[g];
        }
    }

    // h^T state in D layout: hA[j] = h[n = j+8*hi][batch n16],
    //                        hB[j] = h[n = 16+j+8*hi][batch n16]
    v8f hA = {}, hB = {};

    const float2* x2 = reinterpret_cast<const float2*>(x);
    const size_t xbase = (size_t)(b0 + n16) * (size_t)T;
    float2 xv = x2[xbase];   // t = 0

    for (int t = 0; t < T; ++t) {
        // Prefetch next timestep's x while this step computes.
        float2 xnv = x2[xbase + (t + 1 < T ? t + 1 : t)];

        // B operand for f32 WMMA: x^T (2x16); lanes 16-31 carry K=2,3 = 0.
        v2f Bx;
        Bx[0] = hi ? 0.0f : xv.x;
        Bx[1] = hi ? 0.0f : xv.y;

        // B operand for f16 WMMA: h^T (32x16). lane needs K = 16*hi + 0..15.
        // Half-wave exchange supplies the 8 K-values held by the partner lane.
        v16h Bh;
        #pragma unroll
        for (int j = 0; j < 8; ++j) {
            float send = hi ? hA[j] : hB[j];
            float recv = __shfl_xor(send, 16, 32);
            Bh[j]     = (_Float16)(hi ? recv  : hA[j]);  // K = 16*hi + j
            Bh[8 + j] = (_Float16)(hi ? hB[j] : recv);   // K = 16*hi + 8 + j
        }

        // x-side projections (full f32): gx^T tiles with biases in C.
        v8f gx0 = __builtin_amdgcn_wmma_f32_16x16x4_f32(false, Aih[0], false, Bx, (short)0, biasRZ[0], false, false);
        v8f gx1 = __builtin_amdgcn_wmma_f32_16x16x4_f32(false, Aih[1], false, Bx, (short)0, biasRZ[1], false, false);
        v8f gx2 = __builtin_amdgcn_wmma_f32_16x16x4_f32(false, Aih[2], false, Bx, (short)0, biasRZ[2], false, false);
        v8f gx3 = __builtin_amdgcn_wmma_f32_16x16x4_f32(false, Aih[3], false, Bx, (short)0, biasRZ[3], false, false);
        v8f xn0 = __builtin_amdgcn_wmma_f32_16x16x4_f32(false, Aih[4], false, Bx, (short)0, biasXn[0], false, false);
        v8f xn1 = __builtin_amdgcn_wmma_f32_16x16x4_f32(false, Aih[5], false, Bx, (short)0, biasXn[1], false, false);

        // Recurrent projections: r/z tiles accumulate onto gx; hn tiles onto b_hh.
        v8f pr0 = __builtin_amdgcn_wmma_f32_16x16x32_f16(false, Ahh[0], false, Bh, (short)0, gx0, false, false);
        v8f pr1 = __builtin_amdgcn_wmma_f32_16x16x32_f16(false, Ahh[1], false, Bh, (short)0, gx1, false, false);
        v8f pz0 = __builtin_amdgcn_wmma_f32_16x16x32_f16(false, Ahh[2], false, Bh, (short)0, gx2, false, false);
        v8f pz1 = __builtin_amdgcn_wmma_f32_16x16x32_f16(false, Ahh[3], false, Bh, (short)0, gx3, false, false);
        v8f hn0 = __builtin_amdgcn_wmma_f32_16x16x32_f16(false, Ahh[4], false, Bh, (short)0, biasHn[0], false, false);
        v8f hn1 = __builtin_amdgcn_wmma_f32_16x16x32_f16(false, Ahh[5], false, Bh, (short)0, biasHn[1], false, false);

        // Gates + state update; all tiles share (gate-row, batch-col) mapping.
        #pragma unroll
        for (int j = 0; j < 8; ++j) {
            float r0 = fast_sigmoid(pr0[j]);
            float r1 = fast_sigmoid(pr1[j]);
            float z0 = fast_sigmoid(pz0[j]);
            float z1 = fast_sigmoid(pz1[j]);
            float nv0 = fast_tanh(xn0[j] + r0 * hn0[j]);
            float nv1 = fast_tanh(xn1[j] + r1 * hn1[j]);
            hA[j] = nv0 + z0 * (hA[j] - nv0);
            hB[j] = nv1 + z1 * (hB[j] - nv1);
        }

        xv = xnv;
    }

    // ---- Final FC: out[b] = h[b,:] . fc_w + fc_b (reduce over hidden) ----
    float s = 0.0f;
    #pragma unroll
    for (int j = 0; j < 8; ++j) {
        s += hA[j] * fc_w[j + 8 * hi];
        s += hB[j] * fc_w[16 + j + 8 * hi];
    }
    s += __shfl_xor(s, 16, 32);
    if (hi == 0)
        out[b0 + n16] = s + fc_b[0];
}

extern "C" void kernel_launch(void* const* d_in, const int* in_sizes, int n_in,
                              void* d_out, int out_size, void* d_ws, size_t ws_size,
                              hipStream_t stream) {
    const float* x    = (const float*)d_in[0];
    const float* w_ih = (const float*)d_in[1];
    const float* w_hh = (const float*)d_in[2];
    const float* b_ih = (const float*)d_in[3];
    const float* b_hh = (const float*)d_in[4];
    const float* fc_w = (const float*)d_in[5];
    const float* fc_b = (const float*)d_in[6];
    float* out = (float*)d_out;

    const int B = out_size;                       // 8192
    const int T = in_sizes[0] / (B * GRU_I);      // 1024

    dim3 grid(B / 16), block(32);                 // one wave32 per 16-row tile
    hipLaunchKernelGGL(gru_wmma_kernel, grid, block, 0, stream,
                       x, w_ih, w_hh, b_ih, b_hh, fc_w, fc_b, out, T);
}
